// DeltaNet_20401094656183
// MI455X (gfx1250) — matrix operand order repaired and checked by reading, not compiled, not run on hardware
//
#include <hip/hip_runtime.h>

// ---------------------------------------------------------------------------
// Types / helpers for CDNA5 WMMA (gfx1250, wave32)
// ---------------------------------------------------------------------------
typedef __attribute__((ext_vector_type(16))) __bf16 v16bf;
typedef __attribute__((ext_vector_type(8)))  float  v8f;
typedef __attribute__((ext_vector_type(4)))  unsigned int v4u;
typedef __attribute__((ext_vector_type(8)))  int v8i;
typedef __attribute__((ext_vector_type(4)))  int v4i;

union BfFrag { v16bf v; unsigned int u[8]; };
union FFrag  { v8f   v; float        f[8]; };

__device__ __forceinline__ unsigned int f2bf1(float f) {
    unsigned int u = __float_as_uint(f);
    return (u + 0x7fffu + ((u >> 16) & 1u)) >> 16;   // RNE
}
__device__ __forceinline__ unsigned int packbf(float lo, float hi) {
    return (f2bf1(lo) & 0xffffu) | (f2bf1(hi) << 16);
}

__device__ __forceinline__ v8f wmma_bf16(const BfFrag& a, const BfFrag& b, v8f c) {
    return __builtin_amdgcn_wmma_f32_16x16x32_bf16(
        false, a.v, false, b.v, (short)0, c, false, false);
}

// A fragment: 16x32 bf16 tile (MxK); LDS rows pack (k,k+1) pairs per uint.
__device__ __forceinline__ BfFrag load_frag_a(const unsigned int* lds, int stride,
                                              int m0, int k2w, int lane) {
    BfFrag fr;
    const int m  = m0 + (lane & 15);
    const int hh = (lane >> 4) << 2;
    const unsigned int* p = lds + m * stride + k2w;
#pragma unroll
    for (int v = 0; v < 8; ++v)
        fr.u[v] = p[((v >> 2) << 3) + hh + (v & 3)];
    return fr;
}
// B fragment: 32x16 bf16 tile (KxN), stored transposed Bt[n][k2].
__device__ __forceinline__ BfFrag load_frag_b(const unsigned int* lds, int stride,
                                              int n0, int k2w, int lane) {
    BfFrag fr;
    const int n = n0 + (lane & 15);
    const unsigned int* p = lds + n * stride + k2w + ((lane >> 4) << 2);
#pragma unroll
    for (int v = 0; v < 8; ++v) fr.u[v] = p[((v >> 2) << 3) + (v & 3)];
    return fr;
}
// C/D: vgpr r -> (m = m0 + 8*(lane>>4) + r, n = n0 + (lane&15))
__device__ __forceinline__ FFrag load_frag_c(const float* lds, int stride,
                                             int m0, int n0, int lane) {
    FFrag fr;
    const int m = m0 + ((lane >> 4) << 3);
    const int n = n0 + (lane & 15);
#pragma unroll
    for (int r = 0; r < 8; ++r) fr.f[r] = lds[(m + r) * stride + n];
    return fr;
}
__device__ __forceinline__ void store_frag_c(float* lds, int stride,
                                             int m0, int n0, int lane, const FFrag& fr) {
    const int m = m0 + ((lane >> 4) << 3);
    const int n = n0 + (lane & 15);
#pragma unroll
    for (int r = 0; r < 8; ++r) lds[(m + r) * stride + n] = fr.f[r];
}

// ---------------------------------------------------------------------------
// Tensor Data Mover: 2-D tile load  global -> LDS, with LDS row padding.
// dsizeCode: 0=1B,1=2B,2=4B.  padCode: LDS pad of 1 DWORD every 2<<padCode DWORDs.
// ---------------------------------------------------------------------------
__device__ __forceinline__ void tdm_load_2d(unsigned int ldsOff, const void* gaddr,
                                            unsigned int tileRows, unsigned int rowElems,
                                            unsigned int strideElems,
                                            unsigned int dsizeCode, unsigned int padCode) {
    union { v4u v; unsigned int w[4]; } g0;
    size_t ga = (size_t)gaddr;
    g0.w[0] = 1u;                                          // count=1, user mode
    g0.w[1] = ldsOff;                                      // lds_addr (bytes)
    g0.w[2] = (unsigned int)ga;                            // global_addr[31:0]
    g0.w[3] = ((unsigned int)(ga >> 32) & 0x01ffffffu) | (2u << 30);  // addr[56:32], type=2
    union { v8i v; unsigned int w[8]; } g1;
    g1.w[0] = (dsizeCode << 16) | (1u << 20) | (padCode << 22);       // pad_en, pad_amount=1dw
    g1.w[1] = (rowElems & 0xffffu) << 16;                  // tensor_dim0 lo
    g1.w[2] = (rowElems >> 16) | ((tileRows & 0xffffu) << 16);        // dim0 hi | dim1 lo
    g1.w[3] = (tileRows >> 16) | ((rowElems & 0xffffu) << 16);        // dim1 hi | tile_dim0
    g1.w[4] = (tileRows & 0xffffu);                        // tile_dim1 (tile_dim2=0)
    g1.w[5] = strideElems;                                 // tensor_dim0_stride lo
    g1.w[6] = 0;
    g1.w[7] = 0;
    v4i z4 = (v4i)0;
#if defined(__clang_major__) && (__clang_major__ >= 23)
    v8i z8 = (v8i)0;
    __builtin_amdgcn_tensor_load_to_lds(g0.v, g1.v, z4, z4, z8, 0);
#else
    __builtin_amdgcn_tensor_load_to_lds(g0.v, g1.v, z4, z4, 0);
#endif
}

// ---------------------------------------------------------------------------
// fp32 -> bf16 cast (pairs packed into uint)
// ---------------------------------------------------------------------------
__global__ __launch_bounds__(256) void cast_f32_bf16(const float* __restrict__ in,
                                                     unsigned int* __restrict__ out,
                                                     int n2) {
    int i = blockIdx.x * 256 + threadIdx.x;
    if (i < n2) out[i] = packbf(in[2 * i], in[2 * i + 1]);
}

// ---------------------------------------------------------------------------
// bf16 GEMM  C[M,N] = A[M,K] * B[K,N], fp32 accumulate.
// 128x128 WG tile, K-step 32, 8 waves x (64x32), double-buffered LDS.
// ---------------------------------------------------------------------------
__global__ __launch_bounds__(256) void gemm_bf16(const unsigned short* __restrict__ A,
                                                 const unsigned short* __restrict__ Bw,
                                                 float* __restrict__ C,
                                                 int M, int N, int Kd) {
    __shared__ unsigned int lds_a [2 * 128 * 17];
    __shared__ unsigned int lds_bt[2 * 128 * 17];

    const int tid  = threadIdx.x;
    const int lane = tid & 31;
    const int wave = tid >> 5;
    const int m0   = blockIdx.x * 128;
    const int n0   = blockIdx.y * 128;
    const int wm   = (wave & 1) * 64;
    const int wn   = (wave >> 1) * 32;

    FFrag acc[4][2];
#pragma unroll
    for (int mi = 0; mi < 4; ++mi)
#pragma unroll
        for (int ni = 0; ni < 2; ++ni)
#pragma unroll
            for (int r = 0; r < 8; ++r) acc[mi][ni].f[r] = 0.f;

    const int ar0 = tid >> 2,         ach0 = tid & 3;           // A-stage tasks
    const int ar1 = (tid + 256) >> 2, ach1 = (tid + 256) & 3;
    const int bk2 = tid & 15,         bng = tid >> 4;           // B-stage task

    // ---- stage tile 0 into buffer 0 ----
    {
        uint4 x0 = *(const uint4*)(A + (size_t)(m0 + ar0) * Kd + ach0 * 8);
        uint4 x1 = *(const uint4*)(A + (size_t)(m0 + ar1) * Kd + ach1 * 8);
        unsigned int* d0 = lds_a + ar0 * 17 + ach0 * 4;
        unsigned int* d1 = lds_a + ar1 * 17 + ach1 * 4;
        d0[0]=x0.x; d0[1]=x0.y; d0[2]=x0.z; d0[3]=x0.w;
        d1[0]=x1.x; d1[1]=x1.y; d1[2]=x1.z; d1[3]=x1.w;
        uint4 r0 = *(const uint4*)(Bw + (size_t)(2 * bk2)     * N + n0 + bng * 8);
        uint4 r1 = *(const uint4*)(Bw + (size_t)(2 * bk2 + 1) * N + n0 + bng * 8);
        unsigned int a0[4] = { r0.x, r0.y, r0.z, r0.w };
        unsigned int a1[4] = { r1.x, r1.y, r1.z, r1.w };
#pragma unroll
        for (int i = 0; i < 4; ++i) {
            lds_bt[(bng * 8 + 2 * i)     * 17 + bk2] = (a0[i] & 0xffffu) | (a1[i] << 16);
            lds_bt[(bng * 8 + 2 * i + 1) * 17 + bk2] = (a0[i] >> 16) | (a1[i] & 0xffff0000u);
        }
    }
    __syncthreads();

    const int nk = Kd >> 5;
    for (int ks = 0; ks < nk; ++ks) {
        const unsigned int* cA = lds_a  + (ks & 1) * (128 * 17);
        const unsigned int* cB = lds_bt + (ks & 1) * (128 * 17);
        unsigned int* nA = lds_a  + ((ks & 1) ^ 1) * (128 * 17);
        unsigned int* nB = lds_bt + ((ks & 1) ^ 1) * (128 * 17);
        const bool hasNext = (ks + 1) < nk;

        uint4 pa0, pa1, pb0, pb1;
        if (hasNext) {                                  // global loads for next step
            int kt = (ks + 1) << 5;
            pa0 = *(const uint4*)(A + (size_t)(m0 + ar0) * Kd + kt + ach0 * 8);
            pa1 = *(const uint4*)(A + (size_t)(m0 + ar1) * Kd + kt + ach1 * 8);
            pb0 = *(const uint4*)(Bw + (size_t)(kt + 2 * bk2)     * N + n0 + bng * 8);
            pb1 = *(const uint4*)(Bw + (size_t)(kt + 2 * bk2 + 1) * N + n0 + bng * 8);
        }

        BfFrag fa[4], fb[2];
#pragma unroll
        for (int mi = 0; mi < 4; ++mi) fa[mi] = load_frag_a(cA, 17, wm + mi * 16, 0, lane);
#pragma unroll
        for (int ni = 0; ni < 2; ++ni) fb[ni] = load_frag_b(cB, 17, wn + ni * 16, 0, lane);
#pragma unroll
        for (int mi = 0; mi < 4; ++mi)
#pragma unroll
            for (int ni = 0; ni < 2; ++ni)
                acc[mi][ni].v = wmma_bf16(fa[mi], fb[ni], acc[mi][ni].v);

        if (hasNext) {                                  // write staged regs to next buf
            unsigned int* d0 = nA + ar0 * 17 + ach0 * 4;
            unsigned int* d1 = nA + ar1 * 17 + ach1 * 4;
            d0[0]=pa0.x; d0[1]=pa0.y; d0[2]=pa0.z; d0[3]=pa0.w;
            d1[0]=pa1.x; d1[1]=pa1.y; d1[2]=pa1.z; d1[3]=pa1.w;
            unsigned int a0[4] = { pb0.x, pb0.y, pb0.z, pb0.w };
            unsigned int a1[4] = { pb1.x, pb1.y, pb1.z, pb1.w };
#pragma unroll
            for (int i = 0; i < 4; ++i) {
                nB[(bng * 8 + 2 * i)     * 17 + bk2] = (a0[i] & 0xffffu) | (a1[i] << 16);
                nB[(bng * 8 + 2 * i + 1) * 17 + bk2] = (a0[i] >> 16) | (a1[i] & 0xffff0000u);
            }
        }
        __syncthreads();
    }

#pragma unroll
    for (int mi = 0; mi < 4; ++mi)
#pragma unroll
        for (int ni = 0; ni < 2; ++ni) {
            int mt = m0 + wm + mi * 16 + ((lane >> 4) << 3);
            int nn = n0 + wn + ni * 16 + (lane & 15);
#pragma unroll
            for (int r = 0; r < 8; ++r)
                C[(size_t)(mt + r) * N + nn] = acc[mi][ni].f[r];
        }
}

// ---------------------------------------------------------------------------
// causal depthwise conv (K=4) + SiLU, fp32 -> fp32
// ---------------------------------------------------------------------------
__global__ __launch_bounds__(256) void conv_silu(const float* __restrict__ in,
                                                 const float* __restrict__ w,
                                                 float* __restrict__ out, int total) {
    int idx = blockIdx.x * 256 + threadIdx.x;
    if (idx >= total) return;
    int d = idx & 2047;
    int t = (idx >> 11) & 2047;
    float acc = 0.f;
#pragma unroll
    for (int i = 0; i < 4; ++i) {
        int tt = t - 3 + i;
        float xv = (tt >= 0) ? in[idx + (i - 3) * 2048] : 0.f;
        acc = fmaf(w[d * 4 + i], xv, acc);
    }
    out[idx] = acc / (1.f + __expf(-acc));
}

// ---------------------------------------------------------------------------
// per-head L2 norm over 256, optional scale, fp32 -> bf16 (one wave per row)
// ---------------------------------------------------------------------------
__global__ __launch_bounds__(256) void l2norm_bf16(const float* __restrict__ in,
                                                   unsigned int* __restrict__ out,
                                                   float scale, int groups) {
    int lane = threadIdx.x & 31, wave = threadIdx.x >> 5;
    int g = blockIdx.x * 8 + wave;
    if (g >= groups) return;
    const float* p = in + (size_t)g * 256;
    float4 v0 = *(const float4*)(p + lane * 4);
    float4 v1 = *(const float4*)(p + 128 + lane * 4);
    float ss = v0.x*v0.x + v0.y*v0.y + v0.z*v0.z + v0.w*v0.w
             + v1.x*v1.x + v1.y*v1.y + v1.z*v1.z + v1.w*v1.w;
#pragma unroll
    for (int m = 16; m; m >>= 1) ss += __shfl_xor(ss, m, 32);
    float r = rsqrtf(ss + 1e-6f) * scale;
    unsigned int* o = out + (size_t)g * 128;
    o[lane * 2]          = packbf(v0.x * r, v0.y * r);
    o[lane * 2 + 1]      = packbf(v0.z * r, v0.w * r);
    o[64 + lane * 2]     = packbf(v1.x * r, v1.y * r);
    o[64 + lane * 2 + 1] = packbf(v1.z * r, v1.w * r);
}

// ---------------------------------------------------------------------------
// beta = sigmoid(x @ Wb): one wave per row, lane-strided, shfl reduction
// ---------------------------------------------------------------------------
__global__ __launch_bounds__(256) void beta_kernel(const float* __restrict__ x,
                                                   const float* __restrict__ Wb,
                                                   float* __restrict__ beta, int rows) {
    int lane = threadIdx.x & 31, wave = threadIdx.x >> 5;
    int row = blockIdx.x * 8 + wave;
    if (row >= rows) return;
    const float* xr = x + (size_t)row * 2048;
    float acc[8];
#pragma unroll
    for (int h = 0; h < 8; ++h) acc[h] = 0.f;
    for (int d = lane; d < 2048; d += 32) {
        float xv = xr[d];
        float4 w0 = *(const float4*)(Wb + d * 8);
        float4 w1 = *(const float4*)(Wb + d * 8 + 4);
        acc[0] = fmaf(xv, w0.x, acc[0]); acc[1] = fmaf(xv, w0.y, acc[1]);
        acc[2] = fmaf(xv, w0.z, acc[2]); acc[3] = fmaf(xv, w0.w, acc[3]);
        acc[4] = fmaf(xv, w1.x, acc[4]); acc[5] = fmaf(xv, w1.y, acc[5]);
        acc[6] = fmaf(xv, w1.z, acc[6]); acc[7] = fmaf(xv, w1.w, acc[7]);
    }
#pragma unroll
    for (int h = 0; h < 8; ++h) {
        float s = acc[h];
#pragma unroll
        for (int m = 16; m; m >>= 1) s += __shfl_xor(s, m, 32);
        if (lane == 0) beta[row * 8 + h] = 1.f / (1.f + __expf(-s));
    }
}

// ---------------------------------------------------------------------------
// per-head RMS norm over 256 with weight, fp32 -> bf16
// ---------------------------------------------------------------------------
__global__ __launch_bounds__(256) void rmsnorm_bf16(const float* __restrict__ in,
                                                    const float* __restrict__ nw,
                                                    unsigned int* __restrict__ out,
                                                    int groups) {
    int lane = threadIdx.x & 31, wave = threadIdx.x >> 5;
    int g = blockIdx.x * 8 + wave;
    if (g >= groups) return;
    const float* p = in + (size_t)g * 256;
    float4 v0 = *(const float4*)(p + lane * 4);
    float4 v1 = *(const float4*)(p + 128 + lane * 4);
    float ss = v0.x*v0.x + v0.y*v0.y + v0.z*v0.z + v0.w*v0.w
             + v1.x*v1.x + v1.y*v1.y + v1.z*v1.z + v1.w*v1.w;
#pragma unroll
    for (int m = 16; m; m >>= 1) ss += __shfl_xor(ss, m, 32);
    float r = rsqrtf(ss * (1.f / 256.f) + 1e-5f);
    float4 w0 = *(const float4*)(nw + lane * 4);
    float4 w1 = *(const float4*)(nw + 128 + lane * 4);
    unsigned int* o = out + (size_t)g * 128;
    o[lane * 2]          = packbf(v0.x * r * w0.x, v0.y * r * w0.y);
    o[lane * 2 + 1]      = packbf(v0.z * r * w0.z, v0.w * r * w0.w);
    o[64 + lane * 2]     = packbf(v1.x * r * w1.x, v1.y * r * w1.y);
    o[64 + lane * 2 + 1] = packbf(v1.z * r * w1.z, v1.w * r * w1.w);
}

// ---------------------------------------------------------------------------
// Chunked delta rule, one WG per (b, h, dv-slice of 64), chunk C = 32.
// TDM stages K/Q (bf16) and V (f32) chunk tiles into double-buffered LDS,
// prefetching chunk i+1 while chunk i computes.  S fp32 + bf16 mirror in LDS.
// ---------------------------------------------------------------------------
#define KCB_BYTES (32*129*4)
#define VVB_BYTES (32*65*4)
#define OFF_S    0                            // float [256][65]
#define OFF_Sb   (OFF_S    + 256*65*4)        // uint  [64][129]
#define OFF_Kc0  (OFF_Sb   + 64*129*4)        // uint  [32][129] x2
#define OFF_Qc0  (OFF_Kc0  + 2*KCB_BYTES)     // uint  [32][129] x2
#define OFF_Kt   (OFF_Qc0  + 2*KCB_BYTES)     // uint  [256][17]
#define OFF_KS   (OFF_Kt   + 256*17*4)        // float [32][65]
#define OFF_QS   (OFF_KS   + 32*65*4)
#define OFF_Vv0  (OFF_QS   + 32*65*4)         // float [32][65] x2
#define OFF_Uu   (OFF_Vv0  + 2*VVB_BYTES)
#define OFF_Ubt  (OFF_Uu   + 32*65*4)         // uint  [64][17]
#define OFF_Pm   (OFF_Ubt  + 64*17*4)         // uint  [32][17]
#define OFF_AKK  (OFF_Pm   + 32*17*4)         // float [32][33]
#define OFF_Pf   (OFF_AKK  + 32*33*4)
#define OFF_BB   (OFF_Pf   + 32*33*4)         // float [32]
#define DELTA_SMEM (OFF_BB + 32*4)            // ~234 KB < 320 KB WGP LDS

__global__ __launch_bounds__(256) void delta_kernel(const unsigned short* __restrict__ Q,
                                                    const unsigned short* __restrict__ K,
                                                    const float* __restrict__ V,
                                                    const float* __restrict__ Beta,
                                                    float* __restrict__ O, int Tlen) {
    extern __shared__ char smem[];
    float*        S    = (float*)(smem + OFF_S);
    unsigned int* Sb   = (unsigned int*)(smem + OFF_Sb);
    unsigned int* Kt   = (unsigned int*)(smem + OFF_Kt);
    float*        KS   = (float*)(smem + OFF_KS);
    float*        QS   = (float*)(smem + OFF_QS);
    float*        Uu   = (float*)(smem + OFF_Uu);
    unsigned int* Ubt  = (unsigned int*)(smem + OFF_Ubt);
    unsigned int* Pm   = (unsigned int*)(smem + OFF_Pm);
    float*        AKK  = (float*)(smem + OFF_AKK);
    float*        Pf   = (float*)(smem + OFF_Pf);
    float*        BB   = (float*)(smem + OFF_BB);

    const int tid  = threadIdx.x;
    const int lane = tid & 31;
    const int wave = tid >> 5;
    const int slice = blockIdx.x & 3;
    const int h     = (blockIdx.x >> 2) & 7;
    const int b     = blockIdx.x >> 5;

    const size_t baseQK = ((size_t)b * Tlen) * 2048 + h * 256;
    const size_t baseV  = baseQK + slice * 64;

    // issue TDM for chunk 0 while zero-filling state
    if (wave == 0) {
        tdm_load_2d(OFF_Kc0, K + baseQK, 32, 256, 2048, 1, 6);
        tdm_load_2d(OFF_Qc0, Q + baseQK, 32, 256, 2048, 1, 6);
        tdm_load_2d(OFF_Vv0, V + baseV,  32,  64, 2048, 2, 5);
    }
    for (int i = tid; i < 256 * 65; i += 256) S[i] = 0.f;
    for (int i = tid; i < 64 * 129; i += 256) Sb[i] = 0u;

    const int nc = Tlen >> 5;
    for (int ci = 0; ci < nc; ++ci) {
        const int cur = ci & 1;
        const int t0  = ci << 5;
        const unsigned int* Kc = (const unsigned int*)(smem + OFF_Kc0 + cur * KCB_BYTES);
        const unsigned int* Qc = (const unsigned int*)(smem + OFF_Qc0 + cur * KCB_BYTES);
        const float*        Vv = (const float*)(smem + OFF_Vv0 + cur * VVB_BYTES);

        if (tid < 32) BB[tid] = Beta[((size_t)b * Tlen + t0 + tid) * 8 + h];
        if (wave == 0) __builtin_amdgcn_s_wait_tensorcnt(0);   // cur buffers ready
        __syncthreads();
        if (wave == 0 && ci + 1 < nc) {                        // prefetch next chunk
            const int t1 = t0 + 32;
            const int nxt = cur ^ 1;
            tdm_load_2d(OFF_Kc0 + nxt * KCB_BYTES, K + baseQK + (size_t)t1 * 2048, 32, 256, 2048, 1, 6);
            tdm_load_2d(OFF_Qc0 + nxt * KCB_BYTES, Q + baseQK + (size_t)t1 * 2048, 32, 256, 2048, 1, 6);
            tdm_load_2d(OFF_Vv0 + nxt * VVB_BYTES, V + baseV  + (size_t)t1 * 2048, 32,  64, 2048, 2, 5);
        }

        // ---- build K^T pair-packed (for S += K^T @ U) ----
#pragma unroll
        for (int it = 0; it < 16; ++it) {
            int task = tid + it * 256;                         // 256 c x 16 m2
            int c = task >> 4, m2 = task & 15;
            unsigned int w0 = Kc[(2 * m2)     * 129 + (c >> 1)];
            unsigned int w1 = Kc[(2 * m2 + 1) * 129 + (c >> 1)];
            unsigned int lo = (c & 1) ? (w0 >> 16) : (w0 & 0xffffu);
            unsigned int hi = (c & 1) ? (w1 & 0xffff0000u) : (w1 << 16);
            Kt[c * 17 + m2] = lo | hi;
        }
        __syncthreads();

        // ---- phase 1: KS = K@S, QS = Q@S ----
        {
            int tm = wave & 1, tn = wave >> 1;
            FFrag aK, aQ;
#pragma unroll
            for (int r = 0; r < 8; ++r) { aK.f[r] = 0.f; aQ.f[r] = 0.f; }
            for (int kb = 0; kb < 8; ++kb) {
                BfFrag fK = load_frag_a(Kc, 129, tm * 16, kb * 16, lane);
                BfFrag fQ = load_frag_a(Qc, 129, tm * 16, kb * 16, lane);
                BfFrag fS = load_frag_b(Sb, 129, tn * 16, kb * 16, lane);
                aK.v = wmma_bf16(fK, fS, aK.v);
                aQ.v = wmma_bf16(fQ, fS, aQ.v);
            }
            store_frag_c(KS, 65, tm * 16, tn * 16, lane, aK);
            store_frag_c(QS, 65, tm * 16, tn * 16, lane, aQ);
        }
        // ---- phase 2: AKK = K.K^T (waves 0-3), Pf = Q.K^T (waves 4-7) ----
        {
            int w4 = wave & 3, tm = w4 & 1, tn = w4 >> 1;
            const unsigned int* Arow = (wave < 4) ? Kc : Qc;
            FFrag acc;
#pragma unroll
            for (int r = 0; r < 8; ++r) acc.f[r] = 0.f;
            for (int kb = 0; kb < 8; ++kb) {
                BfFrag fa = load_frag_a(Arow, 129, tm * 16, kb * 16, lane);
                BfFrag fb = load_frag_b(Kc, 129, tn * 16, kb * 16, lane);
                acc.v = wmma_bf16(fa, fb, acc.v);
            }
            store_frag_c((wave < 4) ? AKK : Pf, 33, tm * 16, tn * 16, lane, acc);
        }
        __syncthreads();

        // ---- phase 3: serial u recurrence (64 column threads) + masked P ----
        if (tid < 64) {
            int j = tid;
            float u[32];
            for (int t = 0; t < 32; ++t) {
                float acc = Vv[t * 65 + j] - KS[t * 65 + j];
                for (int s = 0; s < t; ++s) acc = fmaf(-AKK[t * 33 + s], u[s], acc);
                u[t] = BB[t] * acc;
            }
#pragma unroll
            for (int t = 0; t < 32; ++t) Uu[t * 65 + j] = u[t];
#pragma unroll
            for (int m2 = 0; m2 < 16; ++m2)
                Ubt[j * 17 + m2] = packbf(u[2 * m2], u[2 * m2 + 1]);
        }
        for (int task = tid; task < 512; task += 256) {        // Pm = tril_incl(P) bf16
            int m = task >> 4, k2 = task & 15;
            float p0 = (2 * k2     <= m) ? Pf[m * 33 + 2 * k2]     : 0.f;
            float p1 = (2 * k2 + 1 <= m) ? Pf[m * 33 + 2 * k2 + 1] : 0.f;
            Pm[m * 17 + k2] = packbf(p0, p1);
        }
        __syncthreads();

        // ---- phase 4: O = QS + Pm @ U  -> global ----
        {
            int tm = wave & 1, tn = wave >> 1;
            FFrag acc = load_frag_c(QS, 65, tm * 16, tn * 16, lane);
            BfFrag fa = load_frag_a(Pm, 17, tm * 16, 0, lane);
            BfFrag fb = load_frag_b(Ubt, 17, tn * 16, 0, lane);
            acc.v = wmma_bf16(fa, fb, acc.v);
            int mr = tm * 16 + ((lane >> 4) << 3);
            int nn = tn * 16 + (lane & 15);
            float* op = O + ((size_t)b * Tlen + t0) * 2048 + h * 256 + slice * 64;
#pragma unroll
            for (int r = 0; r < 8; ++r)
                op[(size_t)(mr + r) * 2048 + nn] = acc.f[r];
        }
        // ---- phase 5: S += K^T @ U (fp32 accum + bf16 mirror) ----
#pragma unroll
        for (int ti = 0; ti < 8; ++ti) {
            int id = wave * 8 + ti, tm = id >> 2, tn = id & 3;
            FFrag acc = load_frag_c(S, 65, tm * 16, tn * 16, lane);
            BfFrag fa = load_frag_a(Kt, 17, tm * 16, 0, lane);
            BfFrag fb = load_frag_b(Ubt, 17, tn * 16, 0, lane);
            acc.v = wmma_bf16(fa, fb, acc.v);
            store_frag_c(S, 65, tm * 16, tn * 16, lane, acc);
            int mr = tm * 16 + ((lane >> 4) << 3);
            int nn = tn * 16 + (lane & 15);
#pragma unroll
            for (int rr = 0; rr < 4; ++rr)
                Sb[nn * 129 + (mr >> 1) + rr] = packbf(acc.f[2 * rr], acc.f[2 * rr + 1]);
        }
        __syncthreads();
    }
}

// ---------------------------------------------------------------------------
extern "C" void kernel_launch(void* const* d_in, const int* in_sizes, int n_in,
                              void* d_out, int out_size, void* d_ws, size_t ws_size,
                              hipStream_t stream) {
    const float* x  = (const float*)d_in[0];
    const float* Wq = (const float*)d_in[1];
    const float* Wk = (const float*)d_in[2];
    const float* Wv = (const float*)d_in[3];
    const float* Wb = (const float*)d_in[4];
    const float* cq = (const float*)d_in[5];
    const float* ck = (const float*)d_in[6];
    const float* cv = (const float*)d_in[7];
    const float* nw = (const float*)d_in[8];
    const float* Wo = (const float*)d_in[9];
    float* out = (float*)d_out;

    const size_t NT = 8192;                 // B*T
    char* ws = (char*)d_ws;
    size_t off = 0;
    auto carve = [&](size_t bytes) -> char* {
        char* p = ws + off;
        off += (bytes + 255) & ~(size_t)255;
        return p;
    };
    unsigned short* xb   = (unsigned short*)carve(NT * 2048 * 2);
    unsigned short* wqb  = (unsigned short*)carve(2048 * 2048 * 2);
    unsigned short* wkb  = (unsigned short*)carve(2048 * 2048 * 2);
    unsigned short* wvb  = (unsigned short*)carve(2048 * 2048 * 2);
    unsigned short* wob  = (unsigned short*)carve(2048 * 2048 * 2);
    float*          proj = (float*)carve(NT * 2048 * 4);
    float*          ctmp = (float*)carve(NT * 2048 * 4);
    float*          vc   = (float*)carve(NT * 2048 * 4);
    unsigned short* qb   = (unsigned short*)carve(NT * 2048 * 2);
    unsigned short* kb   = (unsigned short*)carve(NT * 2048 * 2);
    float*          beta = (float*)carve(NT * 8 * 4);
    float*          ob   = (float*)carve(NT * 2048 * 4);
    unsigned short* onb  = (unsigned short*)carve(NT * 2048 * 2);

    (void)hipFuncSetAttribute(reinterpret_cast<const void*>(delta_kernel),
                              hipFuncAttributeMaxDynamicSharedMemorySize, DELTA_SMEM);

    auto cast = [&](const float* in, void* o, size_t n) {
        int n2 = (int)(n / 2);
        cast_f32_bf16<<<dim3((n2 + 255) / 256), dim3(256), 0, stream>>>(in, (unsigned int*)o, n2);
    };
    cast(x,  xb,  NT * 2048);
    cast(Wq, wqb, 2048 * 2048);
    cast(Wk, wkb, 2048 * 2048);
    cast(Wv, wvb, 2048 * 2048);
    cast(Wo, wob, 2048 * 2048);

    dim3 ggrid(8192 / 128, 2048 / 128);     // 64 x 16
    int cvblocks = (int)(NT * 2048 / 256);
    int nrmblocks = (int)(NT * 8 / 8);      // l2/rms: 8 waves/block

    // Q path
    gemm_bf16<<<ggrid, 256, 0, stream>>>(xb, wqb, proj, 8192, 2048, 2048);
    conv_silu<<<cvblocks, 256, 0, stream>>>(proj, cq, ctmp, (int)(NT * 2048));
    l2norm_bf16<<<nrmblocks, 256, 0, stream>>>(ctmp, (unsigned int*)qb, 0.0625f, (int)(NT * 8));
    // K path
    gemm_bf16<<<ggrid, 256, 0, stream>>>(xb, wkb, proj, 8192, 2048, 2048);
    conv_silu<<<cvblocks, 256, 0, stream>>>(proj, ck, ctmp, (int)(NT * 2048));
    l2norm_bf16<<<nrmblocks, 256, 0, stream>>>(ctmp, (unsigned int*)kb, 1.0f, (int)(NT * 8));
    // V path
    gemm_bf16<<<ggrid, 256, 0, stream>>>(xb, wvb, proj, 8192, 2048, 2048);
    conv_silu<<<cvblocks, 256, 0, stream>>>(proj, cv, vc, (int)(NT * 2048));
    // beta
    beta_kernel<<<dim3(8192 / 8), 256, 0, stream>>>(x, Wb, beta, (int)NT);
    // delta rule (128 WGs = 4 B x 8 H x 4 DV-slices)
    delta_kernel<<<dim3(128), dim3(256), DELTA_SMEM, stream>>>(qb, kb, vc, beta, ob, 2048);
    // RMS norm + output projection
    rmsnorm_bf16<<<nrmblocks, 256, 0, stream>>>(ob, nw, (unsigned int*)onb, (int)(NT * 8));
    gemm_bf16<<<ggrid, 256, 0, stream>>>(onb, wob, out, 8192, 2048, 2048);

    (void)in_sizes; (void)n_in; (void)out_size; (void)ws_size;
}
